// RouterCL4KT_1735166788133
// MI455X (gfx1250) — matrix operand order repaired and checked by reading, not compiled, not run on hardware
//
#include <hip/hip_runtime.h>

// ---------------- model constants ----------------
#define D_    512
#define H_    8
#define HS_   2
#define HD_   6
#define TOPK_ 4
#define L_    2
#define DK_   64
#define NS_   1000
#define B_    8
#define S_    1024
#define FC_   512
#define M_    (B_*S_)   // 8192 rows

typedef __bf16 bf16_t;
typedef __attribute__((ext_vector_type(8)))  __bf16 v8bf;
typedef __attribute__((ext_vector_type(16))) __bf16 v16bf;
typedef __attribute__((ext_vector_type(8)))  float  v8f;

// ---------------- WMMA helpers ----------------
__device__ __forceinline__ v8f wmma_bf16(v16bf a, v16bf b, v8f c) {
  return __builtin_amdgcn_wmma_f32_16x16x32_bf16(false, a, false, b, (short)0, c, false, false);
}

__device__ __forceinline__ v8f zero8() {
  v8f z;
  #pragma unroll
  for (int i = 0; i < 8; ++i) z[i] = 0.f;
  return z;
}

// A fragment: 16x32 bf16 tile of row-major X (leading dim ld), rows m0..m0+15, cols k0..k0+31
// Contiguous k order used here (matches load_B_frag): lanes 0-15 (M=lane): K = e ; lanes 16-31 (M=lane-16): K = 16+e
__device__ __forceinline__ v16bf load_A_frag(const bf16_t* X, int ld, int m0, int k0) {
  int lane = threadIdx.x & 31;
  const bf16_t* p = X + (size_t)(m0 + (lane & 15)) * ld + k0 + ((lane >> 4) ? 16 : 0);
  v8bf lo = *(const v8bf*)p;          // K = kh .. kh+7
  v8bf hi = *(const v8bf*)(p + 8);    // K = kh+8 .. kh+15
  return __builtin_shufflevector(lo, hi, 0,1,2,3,4,5,6,7,8,9,10,11,12,13,14,15);
}

// B fragment: 32x16 where B[k][n] = W[n0+n][k0+k], W row-major with leading dim ld (K-contiguous).
// Contiguous k order used here (matches load_A_frag): lane n (col = n&15): lanes 0-15 hold K=0..15, lanes 16-31 hold K=16..31.
__device__ __forceinline__ v16bf load_B_frag(const bf16_t* W, int ld, int n0, int k0) {
  int lane = threadIdx.x & 31;
  const bf16_t* p = W + (size_t)(n0 + (lane & 15)) * ld + k0 + ((lane >> 4) ? 16 : 0);
  v8bf lo = *(const v8bf*)p;
  v8bf hi = *(const v8bf*)(p + 8);
  return __builtin_shufflevector(lo, hi, 0,1,2,3,4,5,6,7,8,9,10,11,12,13,14,15);
}

// B fragment from a K-major (strided) matrix: B[k][n] = V[(t0+k)*ld + n0+n]
__device__ __forceinline__ v16bf load_B_frag_strided(const bf16_t* V, int ld, int t0, int n0) {
  int lane = threadIdx.x & 31;
  int col = n0 + (lane & 15);
  int kb  = t0 + ((lane >> 4) ? 16 : 0);
  v16bf b;
  #pragma unroll
  for (int e = 0; e < 16; ++e) b[e] = V[(size_t)(kb + e) * ld + col];
  return b;
}

// ---------------- CDNA5 async global->LDS copy ----------------
__device__ __forceinline__ unsigned lds_off_u32(const void* p) {
  // generic -> LDS(addrspace 3) -> 32-bit byte offset from LDS base
  return (unsigned)(unsigned long long)(const __attribute__((address_space(3))) char*)p;
}
__device__ __forceinline__ void async_copy_b128(unsigned ldsoff, const void* gptr) {
  asm volatile("global_load_async_to_lds_b128 %0, %1, off"
               :: "v"(ldsoff), "v"(gptr) : "memory");
}

__device__ __forceinline__ float redadd16(float v) {
  #pragma unroll
  for (int o = 1; o < 16; o <<= 1) v += __shfl_xor(v, o, 16);
  return v;
}
__device__ __forceinline__ float redmax16(float v) {
  #pragma unroll
  for (int o = 1; o < 16; o <<= 1) v = fmaxf(v, __shfl_xor(v, o, 16));
  return v;
}
__device__ __forceinline__ float redadd32(float v) {
  #pragma unroll
  for (int o = 1; o < 32; o <<= 1) v += __shfl_xor(v, o, 32);
  return v;
}

__device__ __forceinline__ float geluf(float x) {
  return 0.5f * x * (1.f + tanhf(0.7978845608028654f * (x + 0.044715f * x * x * x)));
}

// ---------------- utility kernels ----------------
__global__ void zero_kernel(float* p, int n) {
  int i = blockIdx.x * blockDim.x + threadIdx.x;
  if (i < n) p[i] = 0.f;
}

__global__ void cvt_bf16_kernel(const float* __restrict__ src, bf16_t* __restrict__ dst, size_t n) {
  size_t i = (size_t)blockIdx.x * blockDim.x + threadIdx.x;
  if (i < n) dst[i] = (bf16_t)src[i];
}

__global__ void embed_kernel(const int* __restrict__ q, const int* __restrict__ r,
                             const int* __restrict__ pos,
                             const float* __restrict__ qt, const float* __restrict__ it,
                             const float* __restrict__ pt,
                             float* __restrict__ xf, bf16_t* __restrict__ xb,
                             float* __restrict__ yf, bf16_t* __restrict__ yb,
                             float* __restrict__ qe) {
  size_t i = (size_t)blockIdx.x * blockDim.x + threadIdx.x;
  if (i >= (size_t)M_ * D_) return;
  size_t row = i / D_; int d = (int)(i % D_);
  int qi = q[row], ri = r[row], pi = pos[row];
  int mr = (ri > -1) ? ri : 0;
  float qv = qt[(size_t)qi * D_ + d];
  float pv = pt[(size_t)pi * D_ + d];
  float iv = it[((size_t)qi + (size_t)(NS_ + 2) * mr) * D_ + d];
  float xv = qv + pv, yv = iv + pv;
  xf[i] = xv; xb[i] = (bf16_t)xv;
  yf[i] = yv; yb[i] = (bf16_t)yv;
  qe[i] = qv;
}

// ---------------- LDS-tiled bf16 WMMA GEMM with async global->LDS double buffering --------
// C[M,N] = act(A[M,K] @ W[N,K]^T + bias). Block tile 64x64x32, 8 waves (4 m-subtiles x 2
// n-halves); each wave computes two 16x16 tiles. Tiles staged in LDS via
// global_load_async_to_lds_b128 (ASYNCcnt), double buffered.
__global__ __launch_bounds__(256)
void gemm_bf16_kernel(const bf16_t* __restrict__ A, const bf16_t* __restrict__ W,
                      const float* __restrict__ bias, float* __restrict__ Cf,
                      bf16_t* __restrict__ Cb, int M, int N, int K, int act) {
  __shared__ __align__(16) bf16_t sA[2][64 * 32];
  __shared__ __align__(16) bf16_t sB[2][64 * 32];
  int tid  = threadIdx.x;
  int lane = tid & 31;
  int wave = tid >> 5;
  int wm = wave & 3;   // m sub-tile 0..3
  int wn = wave >> 2;  // n 32-col half 0..1
  int m0 = blockIdx.y * 64;
  int n0 = blockIdx.x * 64;

  // per-thread async-copy slot: row 0..63, 16-byte segment 0..3 within the 32-elem k window
  int crow = tid >> 2, cseg = tid & 3;
  const bf16_t* gA = A + (size_t)(m0 + crow) * K + cseg * 8;
  const bf16_t* gB = W + (size_t)(n0 + crow) * K + cseg * 8;
  unsigned lA[2], lB[2];
  lA[0] = lds_off_u32(&sA[0][crow * 32 + cseg * 8]);
  lA[1] = lds_off_u32(&sA[1][crow * 32 + cseg * 8]);
  lB[0] = lds_off_u32(&sB[0][crow * 32 + cseg * 8]);
  lB[1] = lds_off_u32(&sB[1][crow * 32 + cseg * 8]);

  // prefetch k0 = 0 into buffer 0 (2 async ops per wave)
  async_copy_b128(lA[0], gA);
  async_copy_b128(lB[0], gB);

  v8f acc0 = zero8(), acc1 = zero8();
  int buf = 0;
  for (int k0 = 0; k0 < K; k0 += 32) {
    if (k0 + 32 < K) {
      async_copy_b128(lA[buf ^ 1], gA + (k0 + 32));
      async_copy_b128(lB[buf ^ 1], gB + (k0 + 32));
      asm volatile("s_wait_asynccnt 0x2" ::: "memory");  // current buffer's 2 ops done
    } else {
      asm volatile("s_wait_asynccnt 0x0" ::: "memory");
    }
    __syncthreads();  // everyone's slice of buf has landed
    const bf16_t* tA = &sA[buf][0];
    const bf16_t* tB = &sB[buf][0];
    v16bf a  = load_A_frag(tA, 32, wm * 16, 0);
    v16bf b0 = load_B_frag(tB, 32, wn * 32, 0);
    v16bf b1 = load_B_frag(tB, 32, wn * 32 + 16, 0);
    acc0 = wmma_bf16(a, b0, acc0);
    acc1 = wmma_bf16(a, b1, acc1);
    __syncthreads();  // all reads of buf done before it is overwritten next iter
    buf ^= 1;
  }

  int half = lane >> 4;
  #pragma unroll
  for (int t = 0; t < 2; ++t) {
    v8f acc = t ? acc1 : acc0;
    int col = n0 + wn * 32 + t * 16 + (lane & 15);
    float bvv = bias ? bias[col] : 0.f;
    #pragma unroll
    for (int v = 0; v < 8; ++v) {
      int row = m0 + wm * 16 + v + half * 8;
      float x = acc[v] + bvv;
      if (act == 1) x = fmaxf(x, 0.f);
      else if (act == 2) x = geluf(x);
      size_t off = (size_t)row * N + col;
      if (Cf) Cf[off] = x;
      if (Cb) Cb[off] = (bf16_t)x;
    }
  }
}

// ---------------- MoH gating (+ balance-loss partial sums) ----------------
__global__ __launch_bounds__(32)
void gates_kernel(const float* __restrict__ xf, const float* __restrict__ Wg,
                  const float* __restrict__ Wgs, float* __restrict__ gates,
                  float* __restrict__ balacc) {
  int row = blockIdx.x;
  int lane = threadIdx.x;
  const float* xr = xf + (size_t)row * D_;
  float acc[8];
  #pragma unroll
  for (int h = 0; h < 8; ++h) acc[h] = 0.f;
  for (int d = lane; d < D_; d += 32) {
    float xv = xr[d];
    #pragma unroll
    for (int h = 0; h < HD_; ++h) acc[h] += xv * Wg[(size_t)h * D_ + d];
    acc[6] += xv * Wgs[d];
    acc[7] += xv * Wgs[D_ + d];
  }
  #pragma unroll
  for (int h = 0; h < 8; ++h) acc[h] = redadd32(acc[h]);
  if (lane == 0) {
    float dm = acc[0];
    #pragma unroll
    for (int h = 1; h < HD_; ++h) dm = fmaxf(dm, acc[h]);
    float dg[HD_], ds = 0.f;
    #pragma unroll
    for (int h = 0; h < HD_; ++h) { dg[h] = expf(acc[h] - dm); ds += dg[h]; }
    #pragma unroll
    for (int h = 0; h < HD_; ++h) dg[h] /= ds;
    bool keep[HD_];
    #pragma unroll
    for (int h = 0; h < HD_; ++h) keep[h] = false;
    for (int t = 0; t < TOPK_; ++t) {
      int best = -1; float bvv = -1e30f;
      for (int h = 0; h < HD_; ++h)
        if (!keep[h] && dg[h] > bvv) { bvv = dg[h]; best = h; }
      keep[best] = true;
    }
    float sm = fmaxf(acc[6], acc[7]);
    float e0 = expf(acc[6] - sm), e1 = expf(acc[7] - sm);
    float ss = e0 + e1;
    float* grow = gates + (size_t)row * H_;
    grow[0] = e0 / ss; grow[1] = e1 / ss;
    #pragma unroll
    for (int h = 0; h < HD_; ++h) grow[2 + h] = keep[h] ? dg[h] : 0.f;
    #pragma unroll
    for (int h = 0; h < HD_; ++h) {
      atomicAdd(&balacc[h], keep[h] ? 1.f : 0.f);
      atomicAdd(&balacc[HD_ + h], dg[h]);
    }
  }
}

__global__ void bal_finalize_kernel(const float* __restrict__ balacc, float* __restrict__ outbal) {
  if (threadIdx.x == 0 && blockIdx.x == 0) {
    const float inv = 1.f / (float)(B_ * S_);
    float s = 0.f;
    for (int h = 0; h < HD_; ++h) s += (balacc[h] * inv) * (balacc[HD_ + h] * inv);
    atomicAdd(outbal, (float)HD_ * s);
  }
}

// ---------------- fused flash-style attention (one wave per (b,h, 16-query tile)) ----------------
__global__ __launch_bounds__(32)
void attn_kernel(const bf16_t* __restrict__ qb, const bf16_t* __restrict__ kb,
                 const bf16_t* __restrict__ vb, const float* __restrict__ gates,
                 bf16_t* __restrict__ ctxb, int strict) {
  int bh = blockIdx.y;
  int b = bh / H_, h = bh % H_;
  int q0 = blockIdx.x * 16;
  int lane = threadIdx.x & 31;
  int half = lane >> 4;
  const bf16_t* Qb = qb + (size_t)b * S_ * D_ + h * DK_;
  const bf16_t* Kb = kb + (size_t)b * S_ * D_ + h * DK_;
  const bf16_t* Vb = vb + (size_t)b * S_ * D_ + h * DK_;

  v16bf qa0 = load_A_frag(Qb, D_, q0, 0);
  v16bf qa1 = load_A_frag(Qb, D_, q0, 32);

  v8f o0 = zero8(), o1 = zero8(), o2 = zero8(), o3 = zero8();
  float m_i[8], l_i[8];
  #pragma unroll
  for (int v = 0; v < 8; ++v) { m_i[v] = -1e30f; l_i[v] = 0.f; }

  __shared__ __align__(16) bf16_t Pl[16 * 32];

  int tmax = q0 + 16;  // keys needed: key <= q0+15
  for (int t0 = 0; t0 < tmax; t0 += 32) {
    float sv[2][8]; int mk[2][8];
    #pragma unroll
    for (int sub = 0; sub < 2; ++sub) {
      int kt = t0 + sub * 16;
      v8f sc = zero8();
      v16bf bk0 = load_B_frag(Kb, D_, kt, 0);
      v16bf bk1 = load_B_frag(Kb, D_, kt, 32);
      sc = wmma_bf16(qa0, bk0, sc);
      sc = wmma_bf16(qa1, bk1, sc);
      int key = kt + (lane & 15);
      #pragma unroll
      for (int v = 0; v < 8; ++v) {
        int qrow = q0 + v + half * 8;
        int ok = strict ? (key < qrow) : (key <= qrow);
        sv[sub][v] = ok ? sc[v] * 0.125f : -1e30f;   // 1/sqrt(64)
        mk[sub][v] = ok;
      }
    }
    #pragma unroll
    for (int v = 0; v < 8; ++v) {
      float tmx = redmax16(fmaxf(sv[0][v], sv[1][v]));
      float mnew = fmaxf(m_i[v], tmx);
      float fac = __expf(m_i[v] - mnew);
      float p0 = mk[0][v] ? __expf(sv[0][v] - mnew) : 0.f;
      float p1 = mk[1][v] ? __expf(sv[1][v] - mnew) : 0.f;
      float rs = redadd16(p0 + p1);
      l_i[v] = l_i[v] * fac + rs;
      m_i[v] = mnew;
      o0[v] *= fac; o1[v] *= fac; o2[v] *= fac; o3[v] *= fac;
      int prow = v + half * 8;
      Pl[prow * 32 + (lane & 15)]      = (bf16_t)p0;
      Pl[prow * 32 + (lane & 15) + 16] = (bf16_t)p1;
    }
    __syncthreads();
    v16bf pa = load_A_frag((const bf16_t*)Pl, 32, 0, 0);
    v16bf b0 = load_B_frag_strided(Vb, D_, t0, 0);
    v16bf b1 = load_B_frag_strided(Vb, D_, t0, 16);
    v16bf b2 = load_B_frag_strided(Vb, D_, t0, 32);
    v16bf b3 = load_B_frag_strided(Vb, D_, t0, 48);
    o0 = wmma_bf16(pa, b0, o0);
    o1 = wmma_bf16(pa, b1, o1);
    o2 = wmma_bf16(pa, b2, o2);
    o3 = wmma_bf16(pa, b3, o3);
    __syncthreads();
  }

  #pragma unroll
  for (int v = 0; v < 8; ++v) {
    int qrow = q0 + v + half * 8;
    float g = gates[((size_t)b * S_ + qrow) * H_ + h];
    float invl = (l_i[v] > 0.f) ? (g / l_i[v]) : 0.f;
    size_t base = ((size_t)b * S_ + qrow) * D_ + h * DK_ + (lane & 15);
    ctxb[base +  0] = (bf16_t)(o0[v] * invl);
    ctxb[base + 16] = (bf16_t)(o1[v] * invl);
    ctxb[base + 32] = (bf16_t)(o2[v] * invl);
    ctxb[base + 48] = (bf16_t)(o3[v] * invl);
  }
}

// ---------------- LayerNorm(residual + delta) ----------------
__global__ __launch_bounds__(32)
void ln_kernel(const float* __restrict__ resid, const float* __restrict__ add,
               const float* __restrict__ g, const float* __restrict__ beta,
               float* __restrict__ outf, bf16_t* __restrict__ outb) {
  int row = blockIdx.x;
  int lane = threadIdx.x;
  const float* r = resid + (size_t)row * D_;
  const float* a = add + (size_t)row * D_;
  float vals[16];
  float s = 0.f;
  #pragma unroll
  for (int i = 0; i < 16; ++i) {
    int d = lane + i * 32;
    float v = r[d] + a[d];
    vals[i] = v; s += v;
  }
  s = redadd32(s);
  float mean = s / (float)D_;
  float vs = 0.f;
  #pragma unroll
  for (int i = 0; i < 16; ++i) { float dd = vals[i] - mean; vs += dd * dd; }
  vs = redadd32(vs);
  float inv = rsqrtf(vs / (float)D_ + 1e-5f);
  #pragma unroll
  for (int i = 0; i < 16; ++i) {
    int d = lane + i * 32;
    float o = (vals[i] - mean) * inv * g[d] + beta[d];
    outf[(size_t)row * D_ + d] = o;
    outb[(size_t)row * D_ + d] = (bf16_t)o;
  }
}

__global__ void concat_kernel(const float* __restrict__ xf, const float* __restrict__ qe,
                              bf16_t* __restrict__ hb) {
  size_t i = (size_t)blockIdx.x * blockDim.x + threadIdx.x;
  if (i >= (size_t)M_ * 2 * D_) return;
  size_t row = i / (2 * D_); int d = (int)(i % (2 * D_));
  float v = (d < D_) ? xf[row * D_ + d] : qe[row * D_ + (d - D_)];
  hb[i] = (bf16_t)v;
}

__global__ __launch_bounds__(32)
void logits_kernel(const bf16_t* __restrict__ g2, const float* __restrict__ w3,
                   const float* __restrict__ b3, float* __restrict__ out) {
  int row = blockIdx.x;
  int lane = threadIdx.x;
  float s = 0.f;
  for (int d = lane; d < FC_ / 2; d += 32)
    s += (float)g2[(size_t)row * (FC_ / 2) + d] * w3[d];
  s = redadd32(s);
  if (lane == 0) out[row] = 1.f / (1.f + expf(-(s + b3[0])));
}

// ---------------- host orchestration ----------------
extern "C" void kernel_launch(void* const* d_in, const int* in_sizes, int n_in,
                              void* d_out, int out_size, void* d_ws, size_t ws_size,
                              hipStream_t stream) {
  (void)in_sizes; (void)n_in; (void)out_size; (void)ws_size;
  const int*   q    = (const int*)d_in[0];
  const int*   r    = (const int*)d_in[1];
  const int*   pos  = (const int*)d_in[2];
  const float* qt   = (const float*)d_in[3];
  const float* it   = (const float*)d_in[4];
  const float* pt   = (const float*)d_in[5];
  const float* Wq   = (const float*)d_in[6];
  const float* bq   = (const float*)d_in[7];
  const float* Wv   = (const float*)d_in[8];
  const float* bv   = (const float*)d_in[9];
  const float* Wo   = (const float*)d_in[10];
  const float* bo   = (const float*)d_in[11];
  const float* Wg   = (const float*)d_in[12];
  const float* Wgs  = (const float*)d_in[13];
  const float* ln1g = (const float*)d_in[14];
  const float* ln1b = (const float*)d_in[15];
  const float* Wf1  = (const float*)d_in[16];
  const float* bf1  = (const float*)d_in[17];
  const float* Wf2  = (const float*)d_in[18];
  const float* bf2  = (const float*)d_in[19];
  const float* ln2g = (const float*)d_in[20];
  const float* ln2b = (const float*)d_in[21];
  const float* oW1  = (const float*)d_in[22];
  const float* ob1  = (const float*)d_in[23];
  const float* oW2  = (const float*)d_in[24];
  const float* ob2  = (const float*)d_in[25];
  const float* oW3  = (const float*)d_in[26];
  const float* ob3  = (const float*)d_in[27];
  float* out = (float*)d_out;

  char* wsp = (char*)d_ws;
  auto alloc = [&](size_t bytes) -> char* {
    char* p = wsp; wsp += (bytes + 255) & ~(size_t)255; return p;
  };
  const size_t MD = (size_t)M_ * D_;
  float*  xf   = (float*)alloc(MD * 4);
  float*  yf   = (float*)alloc(MD * 4);
  float*  qe   = (float*)alloc(MD * 4);
  float*  x1f  = (float*)alloc(MD * 4);
  float*  tmpf = (float*)alloc(MD * 4);
  bf16_t* xb   = (bf16_t*)alloc(MD * 2);
  bf16_t* yb   = (bf16_t*)alloc(MD * 2);
  bf16_t* x1b  = (bf16_t*)alloc(MD * 2);
  bf16_t* qbuf = (bf16_t*)alloc(MD * 2);   // q/k projection (shared: kq_same, xq==xk)
  bf16_t* vbuf = (bf16_t*)alloc(MD * 2);   // contiguous after qbuf (MD*2 is 256B multiple)
  bf16_t* ctxb = (bf16_t*)alloc(MD * 2);
  bf16_t* f1b  = (bf16_t*)alloc(MD * 2);
  float*  gatesbuf = (float*)alloc((size_t)M_ * H_ * 4);
  float*  balacc   = (float*)alloc(64 * 4);
  bf16_t* WqB  = (bf16_t*)alloc((size_t)6 * D_ * D_ * 2);
  bf16_t* WvB  = (bf16_t*)alloc((size_t)6 * D_ * D_ * 2);
  bf16_t* WoB  = (bf16_t*)alloc((size_t)6 * D_ * D_ * 2);
  bf16_t* Wf1B = (bf16_t*)alloc((size_t)6 * D_ * D_ * 2);
  bf16_t* Wf2B = (bf16_t*)alloc((size_t)6 * D_ * D_ * 2);
  bf16_t* oW1B = (bf16_t*)alloc((size_t)FC_ * 2 * D_ * 2);
  bf16_t* oW2B = (bf16_t*)alloc((size_t)(FC_ / 2) * FC_ * 2);
  // late-phase aliases (safe: q/v/ctx/f1 dead after the transformer stack)
  bf16_t* hb  = qbuf;  // needs MD*2 elems -> spans qbuf+vbuf (contiguous)
  bf16_t* g1b = ctxb;  // [M, 512]
  bf16_t* g2b = f1b;   // [M, 256]

  auto cvt = [&](const float* src, bf16_t* dst, size_t n) {
    cvt_bf16_kernel<<<dim3((unsigned)((n + 255) / 256)), 256, 0, stream>>>(src, dst, n);
  };
  auto gemm = [&](const bf16_t* A, const bf16_t* W, const float* bias, float* Cf, bf16_t* Cb,
                  int M, int N, int K, int act) {
    gemm_bf16_kernel<<<dim3(N / 64, M / 64), 256, 0, stream>>>(A, W, bias, Cf, Cb, M, N, K, act);
  };

  // weight conversions (idempotent, per-call)
  cvt(Wq,  WqB,  (size_t)6 * D_ * D_);
  cvt(Wv,  WvB,  (size_t)6 * D_ * D_);
  cvt(Wo,  WoB,  (size_t)6 * D_ * D_);
  cvt(Wf1, Wf1B, (size_t)6 * D_ * D_);
  cvt(Wf2, Wf2B, (size_t)6 * D_ * D_);
  cvt(oW1, oW1B, (size_t)FC_ * 2 * D_);
  cvt(oW2, oW2B, (size_t)(FC_ / 2) * FC_);

  zero_kernel<<<1, 32, 0, stream>>>(out + M_, 1);  // balance-loss accumulator in output
  embed_kernel<<<(unsigned)((MD + 255) / 256), 256, 0, stream>>>(q, r, pos, qt, it, pt,
                                                                 xf, xb, yf, yb, qe);

  auto run_layer = [&](int e, int l, float* Xf, bf16_t* Xb, bf16_t* Vinb, int strict) {
    size_t wi = (size_t)e * L_ + l;
    const bf16_t* WqL = WqB + wi * D_ * D_;  const float* bqL = bq + wi * D_;
    const bf16_t* WvL = WvB + wi * D_ * D_;  const float* bvL = bv + wi * D_;
    const bf16_t* WoL = WoB + wi * D_ * D_;  const float* boL = bo + wi * D_;
    const bf16_t* W1L = Wf1B + wi * D_ * D_; const float* b1L = bf1 + wi * D_;
    const bf16_t* W2L = Wf2B + wi * D_ * D_; const float* b2L = bf2 + wi * D_;
    // q (== k: kq_same and xq==xk in every call) and v projections
    gemm(Xb,   WqL, bqL, nullptr, qbuf, M_, D_, D_, 0);
    gemm(Vinb, WvL, bvL, nullptr, vbuf, M_, D_, D_, 0);
    // MoH gates + balance loss
    zero_kernel<<<1, 32, 0, stream>>>(balacc, 2 * HD_);
    gates_kernel<<<M_, 32, 0, stream>>>(Xf, Wg + wi * HD_ * D_, Wgs + wi * HS_ * D_,
                                        gatesbuf, balacc);
    bal_finalize_kernel<<<1, 1, 0, stream>>>(balacc, out + M_);
    // fused attention (gated heads -> bf16 context)
    attn_kernel<<<dim3(S_ / 16, B_ * H_), 32, 0, stream>>>(qbuf, qbuf, vbuf, gatesbuf,
                                                           ctxb, strict);
    // output projection + LN1
    gemm(ctxb, WoL, boL, tmpf, nullptr, M_, D_, D_, 0);
    ln_kernel<<<M_, 32, 0, stream>>>(Xf, tmpf, ln1g + wi * D_, ln1b + wi * D_, x1f, x1b);
    // FFN + LN2 (writes back into the stream buffers)
    gemm(x1b, W1L, b1L, nullptr, f1b, M_, D_, D_, 1);
    gemm(f1b, W2L, b2L, tmpf, nullptr, M_, D_, D_, 0);
    ln_kernel<<<M_, 32, 0, stream>>>(x1f, tmpf, ln2g + wi * D_, ln2b + wi * D_, Xf, Xb);
  };

  for (int l = 0; l < L_; ++l) run_layer(0, l, xf, xb, xb, /*strict=*/0);
  for (int l = 0; l < L_; ++l) run_layer(1, l, yf, yb, yb, /*strict=*/0);
  for (int l = 0; l < L_; ++l) run_layer(2, l, xf, xb, yb, /*strict=*/1);

  // output head
  concat_kernel<<<(unsigned)((MD * 2 + 255) / 256), 256, 0, stream>>>(xf, qe, hb);
  gemm(hb,  oW1B, ob1, nullptr, g1b, M_, FC_,     2 * D_, 2);
  gemm(g1b, oW2B, ob2, nullptr, g2b, M_, FC_ / 2, FC_,    2);
  logits_kernel<<<M_, 32, 0, stream>>>(g2b, oW3, ob3, out);
}
